// MultiHeadAttention_16484084483712
// MI455X (gfx1250) — matrix operand order repaired and checked by reading, not compiled, not run on hardware
//
#include <hip/hip_runtime.h>

// ---------------------------------------------------------------------------
// MI455X (gfx1250) multi-head attention with column-axis softmax.
// Matrix ops: v_wmma_f32_16x16x32_bf16 (fp32 accumulate).
// Data movement: global_load_async_to_lds_b64/b128 (ASYNCcnt) for bf16 tiles.
// ---------------------------------------------------------------------------

typedef __attribute__((ext_vector_type(16))) __bf16 v16bf;
typedef __attribute__((ext_vector_type(8)))  __bf16 v8bf;
typedef __attribute__((ext_vector_type(4)))  __bf16 v4bf;
typedef __attribute__((ext_vector_type(8)))  float  v8f;

#define T_SEQ   2048
#define N_HEADS 16
#define HD      64
#define D_MODEL 1024
#define NROWS   4096      // B*T
#define SCALE   0.125f    // 1/sqrt(64)

__device__ __forceinline__ v8f wmma_bf16(v16bf a, v16bf b, v8f c) {
  // (neg_a, A, neg_b, B, c_mod, C, reuse_a, reuse_b)
  return __builtin_amdgcn_wmma_f32_16x16x32_bf16(false, a, false, b, (short)0, c,
                                                 false, false);
}

__device__ __forceinline__ v8f zero8() {
  v8f z = {0.f, 0.f, 0.f, 0.f, 0.f, 0.f, 0.f, 0.f};
  return z;
}

// --- gfx1250 async memory->LDS copies (ASYNCcnt) ---------------------------
// Generic pointers to LDS keep the LDS byte offset in the low 32 bits
// (flat shared aperture: high 32 bits select the aperture).
__device__ __forceinline__ unsigned lds_off(const void* p) {
  return (unsigned)(unsigned long long)p;
}
__device__ __forceinline__ void async_ld_b128(const void* lds, const void* g) {
  asm volatile("global_load_async_to_lds_b128 %0, %1, off"
               :: "v"(lds_off(lds)), "v"(g) : "memory");
}
__device__ __forceinline__ void async_ld_b64(const void* lds, const void* g) {
  asm volatile("global_load_async_to_lds_b64 %0, %1, off"
               :: "v"(lds_off(lds)), "v"(g) : "memory");
}
__device__ __forceinline__ void wait_async0() {
  asm volatile("s_wait_asynccnt 0x0" ::: "memory");
}

// A-matrix 16x32 bf16 fragment (ISA 7.12.2): lane ml holds row M=ml; element
// e maps to K = (e&7) + (e>>3)*16 + 8*hi  ->  two contiguous 8-elem chunks.
__device__ __forceinline__ v16bf ldA(const __bf16* row, int base, int hi) {
  v8bf lo = *(const v8bf*)(row + base + 8 * hi);
  v8bf hh = *(const v8bf*)(row + base + 16 + 8 * hi);
  return __builtin_shufflevector(lo, hh, 0, 1, 2, 3, 4, 5, 6, 7,
                                 8, 9, 10, 11, 12, 13, 14, 15);
}

// B-matrix 32x16 bf16 fragment: lane ml holds col N=ml; element e maps to
// K = (e&7) + (e>>3)*16 + 8*hi (same as A) when tile is staged [N][K].
__device__ __forceinline__ v16bf ldB(const __bf16* row, int base, int hi) {
  v8bf lo = *(const v8bf*)(row + base + 8 * hi);
  v8bf hh = *(const v8bf*)(row + base + 16 + 8 * hi);
  return __builtin_shufflevector(lo, hh, 0, 1, 2, 3, 4, 5, 6, 7,
                                 8, 9, 10, 11, 12, 13, 14, 15);
}

// ---------------------------------------------------------------------------
// GEMM: Y = X @ W + bias.   X:[M,K] (fp32 or bf16), W:[K,N] fp32.
// OUT_PERHEAD: write bf16 to [B,H,T,64] layout; else fp32 to [M,N].
// Block tile 128x64, BK=32, 8 waves laid out 4(M) x 2(N), 32x32 per wave.
// fp32 inputs must round through VALU (cvt to bf16), so staging stays on the
// load+convert path; next-tile global_prefetch_b8 hides some of the latency.
// ---------------------------------------------------------------------------
template <typename InT, bool OUT_PERHEAD>
__global__ __launch_bounds__(256) void gemm_bias_kernel(
    const InT* __restrict__ X, const float* __restrict__ W,
    const float* __restrict__ bias, void* __restrict__ Yv,
    int M, int N, int K) {
  __shared__ alignas(16) __bf16 Xs[128][40];  // [m][k], stride 40 (16B-aligned)
  __shared__ alignas(16) __bf16 Ws[64][40];   // [n][k] (transposed W tile)

  const int tid  = threadIdx.x;
  const int lane = tid & 31;
  const int wave = tid >> 5;
  const int hi   = lane >> 4;
  const int ml   = lane & 15;
  const int m0   = blockIdx.x * 128;
  const int n0   = blockIdx.y * 64;
  const int wm   = (wave >> 1) * 32;
  const int wn   = (wave & 1) * 32;

  v8f acc[2][2];
  acc[0][0] = zero8(); acc[0][1] = zero8();
  acc[1][0] = zero8(); acc[1][1] = zero8();

  for (int k0 = 0; k0 < K; k0 += 32) {
    // ---- stage X tile 128x32 (convert fp32 -> bf16 if needed) ----
    {
      int r = tid >> 1, cb = (tid & 1) * 16;
      if constexpr (sizeof(InT) == 4) {
        const float* src = (const float*)X + (size_t)(m0 + r) * K + k0 + cb;
        if (k0 + 32 < K) __builtin_prefetch(src + 32, 0, 1);
#pragma unroll
        for (int i = 0; i < 16; i += 4) {
          float4 f = *(const float4*)(src + i);
          Xs[r][cb + i + 0] = (__bf16)f.x;
          Xs[r][cb + i + 1] = (__bf16)f.y;
          Xs[r][cb + i + 2] = (__bf16)f.z;
          Xs[r][cb + i + 3] = (__bf16)f.w;
        }
      } else {
        const __bf16* src = (const __bf16*)X + (size_t)(m0 + r) * K + k0 + cb;
        if (k0 + 32 < K) __builtin_prefetch(src + 32, 0, 1);
        *(v8bf*)&Xs[r][cb]     = *(const v8bf*)(src);
        *(v8bf*)&Xs[r][cb + 8] = *(const v8bf*)(src + 8);
      }
    }
    // ---- stage W tile 32x64, transposed into Ws[n][k] ----
    {
      int k = tid >> 3, nb = (tid & 7) * 8;
      const float* src = W + (size_t)(k0 + k) * N + n0 + nb;
      if (k0 + 32 < K) __builtin_prefetch(src + (size_t)32 * N, 0, 1);
#pragma unroll
      for (int i = 0; i < 8; i += 4) {
        float4 f = *(const float4*)(src + i);
        Ws[nb + i + 0][k] = (__bf16)f.x;
        Ws[nb + i + 1][k] = (__bf16)f.y;
        Ws[nb + i + 2][k] = (__bf16)f.z;
        Ws[nb + i + 3][k] = (__bf16)f.w;
      }
    }
    __syncthreads();

    v16bf a[2], b[2];
#pragma unroll
    for (int i = 0; i < 2; i++) a[i] = ldA(&Xs[wm + i * 16 + ml][0], 0, hi);
#pragma unroll
    for (int j = 0; j < 2; j++) b[j] = ldB(&Ws[wn + j * 16 + ml][0], 0, hi);
#pragma unroll
    for (int i = 0; i < 2; i++)
#pragma unroll
      for (int j = 0; j < 2; j++) acc[i][j] = wmma_bf16(a[i], b[j], acc[i][j]);
    __syncthreads();
  }

  // ---- epilogue: bias + store (C/D layout: lane=col, vgpr=row) ----
#pragma unroll
  for (int i = 0; i < 2; i++) {
#pragma unroll
    for (int j = 0; j < 2; j++) {
      int ncol = n0 + wn + j * 16 + ml;
      float bv = bias[ncol];
#pragma unroll
      for (int r = 0; r < 8; r++) {
        int mrow  = m0 + wm + i * 16 + hi * 8 + r;
        float val = acc[i][j][r] + bv;
        if constexpr (OUT_PERHEAD) {
          int bb = mrow >> 11;            // / T_SEQ
          int tt = mrow & (T_SEQ - 1);
          int h  = ncol >> 6;
          int c  = ncol & (HD - 1);
          ((__bf16*)Yv)[(((size_t)(bb * N_HEADS + h) * T_SEQ + tt) << 6) + c] =
              (__bf16)val;
        } else {
          ((float*)Yv)[(size_t)mrow * N + ncol] = val;
        }
      }
    }
  }
}

// ---------------------------------------------------------------------------
// Pass A: colsum[bh,k] = sum_{q>=k} exp(S[q,k]*SCALE).
// Grid (B*H, T/128). Each workgroup owns 128 key columns (K tile resident in
// LDS via async-to-LDS copies); wave w owns 16 columns. Q streams 16 rows at
// a time from the diagonal, also via async-to-LDS.
// ---------------------------------------------------------------------------
__global__ __launch_bounds__(256) void colsum_kernel(
    const __bf16* __restrict__ qh, const __bf16* __restrict__ kh,
    float* __restrict__ colsum) {
  __shared__ alignas(16) __bf16 Ks[128][72];  // [key][hd], stride 72
  __shared__ alignas(16) __bf16 Qs[16][72];

  const int tid  = threadIdx.x;
  const int lane = tid & 31;
  const int wave = tid >> 5;
  const int hi   = lane >> 4;
  const int ml   = lane & 15;
  const int bh   = blockIdx.x;
  const int kbase = blockIdx.y * 128;

  const __bf16* Kg = kh + ((size_t)bh * T_SEQ + kbase) * HD;
  const __bf16* Qg = qh + (size_t)bh * T_SEQ * HD;

  {  // stage K block 128x64 (resident for whole kernel), async memory->LDS
    int r = tid >> 1, cb = (tid & 1) * 32;
    const __bf16* src = Kg + (size_t)r * HD + cb;
#pragma unroll
    for (int i = 0; i < 32; i += 8) async_ld_b128(&Ks[r][cb + i], src + i);
  }
  wait_async0();
  __syncthreads();

  // B = K^T fragments for this wave's 16 columns (hoisted out of q loop)
  const __bf16* krow = &Ks[wave * 16 + ml][0];
  v16bf bk0 = ldB(krow, 0, hi);   // hd 0..31
  v16bf bk1 = ldB(krow, 32, hi);  // hd 32..63

  const int kcol = kbase + wave * 16 + ml;
  float psum = 0.f;

  for (int q0 = kbase; q0 < T_SEQ; q0 += 16) {
    __syncthreads();
    {  // stage Q 16x64 via async copy (8B per thread)
      int r = tid >> 4, cb = (tid & 15) * 4;
      async_ld_b64(&Qs[r][cb], Qg + (size_t)(q0 + r) * HD + cb);
    }
    wait_async0();
    __syncthreads();

    const __bf16* qrow = &Qs[ml][0];
    v16bf a0 = ldA(qrow, 0, hi);
    v16bf a1 = ldA(qrow, 32, hi);
    v8f s = zero8();
    s = wmma_bf16(a0, bk0, s);
    s = wmma_bf16(a1, bk1, s);
#pragma unroll
    for (int r = 0; r < 8; r++) {
      int qq = q0 + hi * 8 + r;
      psum += (qq >= kcol) ? __expf(s[r] * SCALE) : 0.f;
    }
  }
  psum += __shfl_xor(psum, 16);
  if (hi == 0) colsum[(size_t)bh * T_SEQ + kcol] = psum;
}

// ---------------------------------------------------------------------------
// Pass B: O[q,:] = sum_k exp(S[q,k]*SCALE)/colsum[k] * V[k,:].
// Grid (B*H, T/128). Wave w owns 16 q rows; all waves walk key tiles of 32.
// Q/K tiles staged with async-to-LDS; V staged transposed through VGPRs.
// P is relaid out through LDS (C-layout -> A-layout) and fed to P@V WMMAs.
// Output written bf16 to [B,T,H,64] for the final projection GEMM.
// ---------------------------------------------------------------------------
__global__ __launch_bounds__(256) void attn_out_kernel(
    const __bf16* __restrict__ qh, const __bf16* __restrict__ kh,
    const __bf16* __restrict__ vh, const float* __restrict__ colsum,
    __bf16* __restrict__ attn) {
  __shared__ alignas(16) __bf16 Qs[128][72];    // [q][hd]
  __shared__ alignas(16) __bf16 Kt[32][72];     // [key][hd]
  __shared__ alignas(16) __bf16 Vt[64][40];     // [hd][key] (transposed)
  __shared__ alignas(16) __bf16 Ps[8][16][40];  // per-wave P relayout slab

  const int tid  = threadIdx.x;
  const int lane = tid & 31;
  const int wave = tid >> 5;
  const int hi   = lane >> 4;
  const int ml   = lane & 15;
  const int bh   = blockIdx.x;
  const int qbase = blockIdx.y * 128;

  const __bf16* Qg = qh + ((size_t)bh * T_SEQ + qbase) * HD;
  const __bf16* Kg = kh + (size_t)bh * T_SEQ * HD;
  const __bf16* Vg = vh + (size_t)bh * T_SEQ * HD;
  const float*  Cs = colsum + (size_t)bh * T_SEQ;

  {  // stage Q block 128x64 (resident), async memory->LDS
    int r = tid >> 1, cb = (tid & 1) * 32;
    const __bf16* src = Qg + (size_t)r * HD + cb;
#pragma unroll
    for (int i = 0; i < 32; i += 8) async_ld_b128(&Qs[r][cb + i], src + i);
  }
  wait_async0();
  __syncthreads();

  const __bf16* qrow = &Qs[wave * 16 + ml][0];
  v16bf aq0 = ldA(qrow, 0, hi);
  v16bf aq1 = ldA(qrow, 32, hi);

  v8f o[4];
  o[0] = zero8(); o[1] = zero8(); o[2] = zero8(); o[3] = zero8();

  for (int k0 = 0; k0 < qbase + 128; k0 += 32) {
    __syncthreads();
    {  // stage K tile 32x64 [key][hd], async (16B per thread)
      int r = tid >> 3, cb = (tid & 7) * 8;
      async_ld_b128(&Kt[r][cb], Kg + (size_t)(k0 + r) * HD + cb);
    }
    {  // stage V tile transposed -> Vt[hd][key] (needs VGPR bounce)
      int r = tid >> 3, cb = (tid & 7) * 8;
      const __bf16* src = Vg + (size_t)(k0 + r) * HD + cb;
      v8bf vv = *(const v8bf*)src;
#pragma unroll
      for (int i = 0; i < 8; i++) Vt[cb + i][r] = vv[i];
    }
    wait_async0();
    __syncthreads();

    // scores for this wave's 16 q rows x 32 keys; write P in A-layout slab
#pragma unroll
    for (int g = 0; g < 2; g++) {
      const __bf16* krow = &Kt[g * 16 + ml][0];
      v16bf bk0 = ldB(krow, 0, hi);
      v16bf bk1 = ldB(krow, 32, hi);
      v8f s = zero8();
      s = wmma_bf16(aq0, bk0, s);   // same accumulation order as colsum pass
      s = wmma_bf16(aq1, bk1, s);
      int kc = k0 + g * 16 + ml;
      float rinv = 1.0f / Cs[kc];
#pragma unroll
      for (int r = 0; r < 8; r++) {
        int qq = qbase + wave * 16 + hi * 8 + r;
        float p = (qq >= kc) ? __expf(s[r] * SCALE) * rinv : 0.f;
        Ps[wave][hi * 8 + r][g * 16 + ml] = (__bf16)p;
      }
    }
    __syncthreads();  // make P slab visible before A-layout reads

    v16bf pa = ldA(&Ps[wave][ml][0], 0, hi);
#pragma unroll
    for (int c = 0; c < 4; c++) {
      v16bf bv = ldB(&Vt[c * 16 + ml][0], 0, hi);  // B[key][hd-col]
      o[c] = wmma_bf16(pa, bv, o[c]);
    }
  }

  // write O -> attn[B,T,H,64] (bf16) for the output projection
  int b = bh >> 4, h = bh & (N_HEADS - 1);
#pragma unroll
  for (int c = 0; c < 4; c++) {
    int hd = c * 16 + ml;
#pragma unroll
    for (int r = 0; r < 8; r++) {
      int t = qbase + wave * 16 + hi * 8 + r;
      attn[(((size_t)b * T_SEQ + t) * N_HEADS + h) * HD + hd] = (__bf16)o[c][r];
    }
  }
}

// ---------------------------------------------------------------------------
// Host launcher
// ---------------------------------------------------------------------------
extern "C" void kernel_launch(void* const* d_in, const int* in_sizes, int n_in,
                              void* d_out, int out_size, void* d_ws,
                              size_t ws_size, hipStream_t stream) {
  const float* q   = (const float*)d_in[0];
  const float* k   = (const float*)d_in[1];
  const float* v   = (const float*)d_in[2];
  const float* w_q = (const float*)d_in[3];
  const float* b_q = (const float*)d_in[4];
  const float* w_k = (const float*)d_in[5];
  const float* b_k = (const float*)d_in[6];
  const float* w_v = (const float*)d_in[7];
  const float* b_v = (const float*)d_in[8];
  const float* w_o = (const float*)d_in[9];
  const float* b_o = (const float*)d_in[10];
  float* out = (float*)d_out;

  // workspace layout (bf16 per-head Q/K/V: 8MB each; colsum 256KB; attn 8MB)
  const size_t NH = (size_t)2 * N_HEADS * T_SEQ * HD;  // 4M elems per tensor
  __bf16* qh = (__bf16*)d_ws;
  __bf16* kh = qh + NH;
  __bf16* vh = kh + NH;
  float* colsum = (float*)(vh + NH);
  __bf16* attn = (__bf16*)(colsum + (size_t)2 * N_HEADS * T_SEQ);

  dim3 gg(NROWS / 128, D_MODEL / 64);  // (32,16)
  dim3 ga(2 * N_HEADS, T_SEQ / 128);   // (32,16)

  gemm_bias_kernel<float, true><<<gg, 256, 0, stream>>>(
      q, w_q, b_q, qh, NROWS, D_MODEL, D_MODEL);
  gemm_bias_kernel<float, true><<<gg, 256, 0, stream>>>(
      k, w_k, b_k, kh, NROWS, D_MODEL, D_MODEL);
  gemm_bias_kernel<float, true><<<gg, 256, 0, stream>>>(
      v, w_v, b_v, vh, NROWS, D_MODEL, D_MODEL);
  colsum_kernel<<<ga, 256, 0, stream>>>(qh, kh, colsum);
  attn_out_kernel<<<ga, 256, 0, stream>>>(qh, kh, vh, colsum, attn);
  gemm_bias_kernel<__bf16, false><<<gg, 256, 0, stream>>>(
      attn, w_o, b_o, out, NROWS, D_MODEL, D_MODEL);
}
